// DiffPool4GraphLayer_50646254354502
// MI455X (gfx1250) — matrix operand order repaired and checked by reading, not compiled, not run on hardware
//
#include <hip/hip_runtime.h>
#include <hip/hip_bf16.h>

// ---------------- problem constants (match reference) ----------------
#define N_NODES    16000
#define IN_DIM     256
#define K_DIM      512
#define FEAT_DIM   256
#define ASSIGN_DIM 1600
#define NPG        500     // nodes per graph
#define CPG        50      // clusters per graph
#define NGRAPH     32

typedef __attribute__((ext_vector_type(16))) __bf16 v16bf;
typedef __attribute__((ext_vector_type(8)))  float  v8f;

// ---------------- utility ----------------
__global__ void zero_f32(float* __restrict__ p, int n) {
  int i = blockIdx.x * blockDim.x + threadIdx.x;
  if (i < n) p[i] = 0.0f;
}

// ---------------- edge aggregation: msg_sum / deg ----------------
__global__ void agg_msg(const float* __restrict__ h, const int* __restrict__ src,
                        const int* __restrict__ dst, float* __restrict__ msg, int E) {
  int idx = blockIdx.x * blockDim.x + threadIdx.x;   // E * 64 (float4 chunks)
  if (idx >= E * 64) return;
  int e = idx >> 6, f4 = idx & 63;
  int s = src[e], d = dst[e];
  const float4 v = *reinterpret_cast<const float4*>(h + (size_t)s * IN_DIM + f4 * 4);
  float* o = msg + (size_t)d * IN_DIM + f4 * 4;
  atomicAdd(o + 0, v.x); atomicAdd(o + 1, v.y);
  atomicAdd(o + 2, v.z); atomicAdd(o + 3, v.w);
}

__global__ void agg_deg(const int* __restrict__ dst, float* __restrict__ deg, int E) {
  int e = blockIdx.x * blockDim.x + threadIdx.x;
  if (e < E) atomicAdd(&deg[dst[e]], 1.0f);
}

__global__ void mean_div(float* __restrict__ msg, const float* __restrict__ deg, int n4) {
  int idx = blockIdx.x * blockDim.x + threadIdx.x;   // N * 64
  if (idx >= n4) return;
  float inv = 1.0f / fmaxf(deg[idx >> 6], 1.0f);
  float4* p = reinterpret_cast<float4*>(msg) + idx;
  float4 v = *p;
  v.x *= inv; v.y *= inv; v.z *= inv; v.w *= inv;
  *p = v;
}

// ---------------- WMMA GEMM: bundle = [h | c] @ W + b ----------------
// A = concat(h, c): [16000 x 512] f32 -> bf16 fragments (f32 accumulate)
// B = W: [512 x NOUT] f32 -> bf16 staged in LDS (transposed, padded)
// 8 waves/block; each wave computes a 16x64 tile (4 WMMAs reuse one A fragment).
// Block footprint: 128 rows x 64 cols.
__global__ __launch_bounds__(256) void gemm_bundle(
    const float* __restrict__ h, const float* __restrict__ c,
    const float* __restrict__ W, const float* __restrict__ bias,
    float* __restrict__ out, int NOUT) {
  __shared__ __bf16 Bs[64 * 40];           // [n][k], padded stride 40 elems (80B, conflict-free)
  const int lane = threadIdx.x & 31;
  const int wave = threadIdx.x >> 5;
  const int half = lane >> 4;
  const int ln   = lane & 15;
  const int rowBase = blockIdx.y * 128 + wave * 16;
  const int colBase = blockIdx.x * 64;
  const int m = rowBase + ln;              // A row for this lane

  v8f acc[4] = {v8f{}, v8f{}, v8f{}, v8f{}};

  const int sn  = threadIdx.x & 63;        // staging: column
  const int skp = threadIdx.x >> 6;        // staging: k-pair group 0..3

  for (int kb = 0; kb < K_DIM; kb += 32) {
    // ---- stage B tile [32 k][64 n] into LDS (transposed), f32->bf16 packed ----
    __syncthreads();
    #pragma unroll
    for (int q = 0; q < 4; q++) {
      int k0 = 2 * (skp + 4 * q);          // even k in 0..30
      float w0 = W[(size_t)(kb + k0)     * NOUT + colBase + sn];
      float w1 = W[(size_t)(kb + k0 + 1) * NOUT + colBase + sn];
      union { __bf16 hh[2]; unsigned u; } pk;
      pk.hh[0] = (__bf16)w0; pk.hh[1] = (__bf16)w1;
      *reinterpret_cast<unsigned*>(&Bs[sn * 40 + k0]) = pk.u;   // 4B-aligned
    }
    __syncthreads();

    // ---- A fragment: lane holds row m, K run {k1..k1+15} ----
    const float* Aptr;
    int k1 = kb + half * 16;
    if (kb < IN_DIM) { Aptr = h + (size_t)m * IN_DIM; }
    else             { Aptr = c + (size_t)m * IN_DIM; k1 -= IN_DIM; }
    __builtin_prefetch(Aptr + k1 + 32, 0, 1);   // global_prefetch_b8 next K tile
    float4 a0 = *reinterpret_cast<const float4*>(Aptr + k1);
    float4 a1 = *reinterpret_cast<const float4*>(Aptr + k1 + 4);
    float4 a2 = *reinterpret_cast<const float4*>(Aptr + k1 + 8);
    float4 a3 = *reinterpret_cast<const float4*>(Aptr + k1 + 12);
    v16bf afrag;
    afrag[0]  = (__bf16)a0.x; afrag[1]  = (__bf16)a0.y;
    afrag[2]  = (__bf16)a0.z; afrag[3]  = (__bf16)a0.w;
    afrag[4]  = (__bf16)a1.x; afrag[5]  = (__bf16)a1.y;
    afrag[6]  = (__bf16)a1.z; afrag[7]  = (__bf16)a1.w;
    afrag[8]  = (__bf16)a2.x; afrag[9]  = (__bf16)a2.y;
    afrag[10] = (__bf16)a2.z; afrag[11] = (__bf16)a2.w;
    afrag[12] = (__bf16)a3.x; afrag[13] = (__bf16)a3.y;
    afrag[14] = (__bf16)a3.z; afrag[15] = (__bf16)a3.w;

    // ---- 4 B fragments (cols ln, ln+16, ln+32, ln+48), one WMMA each ----
    #pragma unroll
    for (int t = 0; t < 4; t++) {
      struct B32 { uint4 a, b; };
      B32 btmp = *reinterpret_cast<const B32*>(&Bs[(t * 16 + ln) * 40 + half * 16]);
      v16bf bfrag = __builtin_bit_cast(v16bf, btmp);
      acc[t] = __builtin_amdgcn_wmma_f32_16x16x32_bf16(
          false, afrag, false, bfrag, (short)0, acc[t], false, false);
    }
  }

  #pragma unroll
  for (int t = 0; t < 4; t++) {
    int col = colBase + t * 16 + ln;
    float bv = bias[col];
    #pragma unroll
    for (int r = 0; r < 8; r++) {
      out[(size_t)(rowBase + half * 8 + r) * NOUT + col] = acc[t][r] + bv;
    }
  }
}

// ---------------- feat: l2-normalize row (256) + relu, in place ----------------
__global__ __launch_bounds__(256) void norm_relu_feat(float* __restrict__ b) {
  int row  = blockIdx.x * 8 + (threadIdx.x >> 5);
  int lane = threadIdx.x & 31;
  float* r = b + (size_t)row * FEAT_DIM;
  float v[8]; float ss = 0.0f;
  #pragma unroll
  for (int t = 0; t < 8; t++) { v[t] = r[lane + 32 * t]; ss += v[t] * v[t]; }
  #pragma unroll
  for (int o = 16; o > 0; o >>= 1) ss += __shfl_xor(ss, o, 32);
  float inv = 1.0f / fmaxf(sqrtf(ss), 1e-12f);
  #pragma unroll
  for (int t = 0; t < 8; t++) r[lane + 32 * t] = fmaxf(v[t] * inv, 0.0f);
}

// ---------------- assign: row norm + relu + masked softmax -> compact s ----------------
// Reference: s = softmax(assign*mask)*mask; s /= (sum(s)+1e-13). Since assign>=0 after
// relu and masked-out columns are 0, this equals e_j / (S_in + 1e-13*(S_in + 1550*exp(-m))).
__global__ __launch_bounds__(256) void softmax_assign(const float* __restrict__ bundle,
                                                      float* __restrict__ s_block) {
  int row  = blockIdx.x * 8 + (threadIdx.x >> 5);
  int lane = threadIdx.x & 31;
  const float* r = bundle + (size_t)row * ASSIGN_DIM;
  float ss = 0.0f;
  for (int t = 0; t < ASSIGN_DIM / 32; t++) { float v = r[lane + 32 * t]; ss += v * v; }
  #pragma unroll
  for (int o = 16; o > 0; o >>= 1) ss += __shfl_xor(ss, o, 32);
  float inv = 1.0f / fmaxf(sqrtf(ss), 1e-12f);

  int g = row / NPG;
  int base = g * CPG;
  bool hasb = (lane + 32) < CPG;
  float za = fmaxf(r[base + lane] * inv, 0.0f);
  float zb = hasb ? fmaxf(r[base + lane + 32] * inv, 0.0f) : 0.0f;
  float m = fmaxf(za, zb);
  #pragma unroll
  for (int o = 16; o > 0; o >>= 1) m = fmaxf(m, __shfl_xor(m, o, 32));
  float ea = __expf(za - m);
  float eb = hasb ? __expf(zb - m) : 0.0f;
  float S = ea + eb;
  #pragma unroll
  for (int o = 16; o > 0; o >>= 1) S += __shfl_xor(S, o, 32);
  float D = S + (float)(ASSIGN_DIM - CPG) * __expf(-m);
  float invd = 1.0f / (S + 1e-13f * D);
  s_block[(size_t)row * CPG + lane] = ea * invd;
  if (hasb) s_block[(size_t)row * CPG + lane + 32] = eb * invd;
}

// ---------------- a_s[v] += s[u] over edges (block-sparse s, 50 cols/edge) ----------------
__global__ void as_scatter(const int* __restrict__ src, const int* __restrict__ dst,
                           const float* __restrict__ s_block, float* __restrict__ a_s, int E) {
  int idx = blockIdx.x * blockDim.x + threadIdx.x;   // E * 50
  if (idx >= E * CPG) return;
  int e = idx / CPG, j = idx - e * CPG;
  int u = src[e], v = dst[e];
  float sv = s_block[(size_t)u * CPG + j];
  int col = (u / NPG) * CPG + j;
  atomicAdd(&a_s[(size_t)v * ASSIGN_DIM + col], sv);
}

// ---------------- h_pool[g*50+cl][f] = sum_{v in g} s[v][cl] * feat[v][f] ----------------
__global__ __launch_bounds__(256) void hpool_kernel(const float* __restrict__ feat,
                                                    const float* __restrict__ s_block,
                                                    float* __restrict__ out) {
  int p = blockIdx.x;               // 0..1599
  int g = p / CPG, cl = p - g * CPG;
  int f = threadIdx.x;              // 0..255
  const int vbase = g * NPG;
  float acc = 0.0f;
  for (int v = 0; v < NPG; v++) {
    float s = s_block[(size_t)(vbase + v) * CPG + cl];
    acc += s * feat[(size_t)(vbase + v) * FEAT_DIM + f];
  }
  out[(size_t)p * FEAT_DIM + f] = acc;
}

// ---------------- adj[g*50+j][q] = sum_{v in g} s[v][j] * a_s[v][q] ----------------
// Each thread owns one q column and all 50 cluster accumulators -> a_s read exactly once.
__global__ __launch_bounds__(256) void adj_kernel(const float* __restrict__ a_s,
                                                  const float* __restrict__ s_block,
                                                  float* __restrict__ out) {
  int g = blockIdx.y;
  int q = blockIdx.x * 256 + threadIdx.x;
  bool act = q < ASSIGN_DIM;
  const int vbase = g * NPG;
  float acc[CPG];
  #pragma unroll
  for (int j = 0; j < CPG; j++) acc[j] = 0.0f;
  for (int v = 0; v < NPG; v++) {
    float av = act ? a_s[(size_t)(vbase + v) * ASSIGN_DIM + q] : 0.0f;
    const float* sp = s_block + (size_t)(vbase + v) * CPG;
    #pragma unroll
    for (int j = 0; j < CPG; j++) acc[j] += sp[j] * av;
  }
  if (act) {
    #pragma unroll
    for (int j = 0; j < CPG; j++)
      out[(size_t)(g * CPG + j) * ASSIGN_DIM + q] = acc[j];
  }
}

// ---------------- launch ----------------
extern "C" void kernel_launch(void* const* d_in, const int* in_sizes, int n_in,
                              void* d_out, int out_size, void* d_ws, size_t ws_size,
                              hipStream_t stream) {
  (void)n_in; (void)out_size; (void)ws_size;
  const float* h      = (const float*)d_in[0];
  const float* W_feat = (const float*)d_in[1];
  const float* b_feat = (const float*)d_in[2];
  const float* W_pool = (const float*)d_in[3];
  const float* b_pool = (const float*)d_in[4];
  const int*   src    = (const int*)d_in[5];
  const int*   dst    = (const int*)d_in[6];
  const int E = in_sizes[5];

  // workspace layout (floats):
  float* ws   = (float*)d_ws;
  float* msg  = ws;                                   // N*256  (msg_sum -> c, in place)
  float* deg  = msg + (size_t)N_NODES * IN_DIM;       // N
  float* feat = deg + N_NODES;                        // N*256  (bundle_feat -> feat)
  float* sblk = feat + (size_t)N_NODES * FEAT_DIM;    // N*50   (compact s)
  float* asg  = sblk + (size_t)N_NODES * CPG;         // N*1600 (bundle_assign -> a_s)

  float* adj_out   = (float*)d_out;                               // 1600*1600
  float* hpool_out = adj_out + (size_t)ASSIGN_DIM * ASSIGN_DIM;   // 1600*256

  // 1) zero msg+deg (contiguous)
  { int n = N_NODES * IN_DIM + N_NODES;
    zero_f32<<<(n + 255) / 256, 256, 0, stream>>>(msg, n); }
  // 2) mean aggregation
  agg_msg<<<(E * 64 + 255) / 256, 256, 0, stream>>>(h, src, dst, msg, E);
  agg_deg<<<(E + 255) / 256, 256, 0, stream>>>(dst, deg, E);
  mean_div<<<(N_NODES * 64 + 255) / 256, 256, 0, stream>>>(msg, deg, N_NODES * 64);
  // 3) WMMA GEMMs (bf16 inputs, f32 accumulate); 64-wide column tiles
  gemm_bundle<<<dim3(FEAT_DIM / 64, N_NODES / 128), 256, 0, stream>>>(
      h, msg, W_feat, b_feat, feat, FEAT_DIM);
  gemm_bundle<<<dim3(ASSIGN_DIM / 64, N_NODES / 128), 256, 0, stream>>>(
      h, msg, W_pool, b_pool, asg, ASSIGN_DIM);
  // 4) normalize + relu / masked softmax
  norm_relu_feat<<<N_NODES / 8, 256, 0, stream>>>(feat);
  softmax_assign<<<N_NODES / 8, 256, 0, stream>>>(asg, sblk);
  // 5) reuse bundle_assign buffer as a_s
  { int n = N_NODES * ASSIGN_DIM;
    zero_f32<<<(n + 255) / 256, 256, 0, stream>>>(asg, n); }
  as_scatter<<<(E * CPG + 255) / 256, 256, 0, stream>>>(src, dst, sblk, asg, E);
  // 6) outputs
  hpool_kernel<<<ASSIGN_DIM, 256, 0, stream>>>(feat, sblk, hpool_out);
  adj_kernel<<<dim3((ASSIGN_DIM + 255) / 256, NGRAPH), 256, 0, stream>>>(asg, sblk, adj_out);
}
